// Encode_Block_34437047780051
// MI455X (gfx1250) — matrix operand order repaired and checked by reading, not compiled, not run on hardware
//
#include <hip/hip_runtime.h>
#include <hip/hip_bf16.h>

#define N_NODES   10000
#define F_IN      512
#define F_E       128
#define F_OUT     128
#define NUM_G     64
#define T_STEPS   3
#define LN_EPS    1e-5f
#define HASH_BITS 19
#define HASH_SZ   (1 << HASH_BITS)
#define HASH_MASK (HASH_SZ - 1)

typedef __bf16 bf16_t;
typedef bf16_t v16bf __attribute__((ext_vector_type(16)));
typedef float  v8f   __attribute__((ext_vector_type(8)));

enum { ACT_NONE = 0, ACT_RELU = 1, ACT_ELU = 2 };

__device__ __forceinline__ float act_apply(float v, int act) {
  if (act == ACT_RELU) return v > 0.f ? v : 0.f;
  if (act == ACT_ELU)  return v > 0.f ? v : (__expf(v) - 1.f);
  return v;
}

// ---------------------------------------------------------------------------
// Weight pre-pack: f32 [K,N] row-major -> bf16 WMMA B-fragment layout.
// Fragment for (n16, k32): 512 bf16, indexed [lane][j] with
//   krow = k32*32 + (lane>>4)*8 + j + (j>=8 ? 8 : 0),  ncol = n16*16 + (lane&15)
// stored at P[((n16*(K/32) + k32)*32 + lane)*16 + j]  (32B contiguous per lane).
// ---------------------------------------------------------------------------
__global__ void pack_b_kernel(const float* __restrict__ B, bf16_t* __restrict__ P,
                              int K, int N)
{
  int idx = blockIdx.x * blockDim.x + threadIdx.x;   // enumerates (n16, k32, lane)
  int total = (N >> 4) * (K >> 5) * 32;
  if (idx >= total) return;
  int lane = idx & 31;
  int t    = idx >> 5;
  int kt32 = K >> 5;
  int k32  = t % kt32;
  int n16  = t / kt32;
  int krow = k32 * 32 + (lane >> 4) * 8;
  int ncol = n16 * 16 + (lane & 15);
  bf16_t* out = P + (size_t)idx * 16;
  #pragma unroll
  for (int j = 0; j < 16; ++j)
    out[j] = (bf16_t)B[(size_t)(krow + j + (j >= 8 ? 8 : 0)) * N + ncol];
}

// ---------------------------------------------------------------------------
// A-fragment 8-element loaders (k0 multiple of 8; each 8-group stays within
// one source because all K's are multiples of 32).
// ---------------------------------------------------------------------------
__device__ __forceinline__ void load8_concat(float out[8], int k0,
    const float* __restrict__ A0, int r0, int K0,
    const float* __restrict__ A1, int r1, int K1)
{
  const float* p;
  if (k0 < K0) p = A0 + (size_t)r0 * K0 + k0;
  else         p = A1 + (size_t)r1 * K1 + (k0 - K0);
  float4 q0 = ((const float4*)p)[0];
  float4 q1 = ((const float4*)p)[1];
  out[0]=q0.x; out[1]=q0.y; out[2]=q0.z; out[3]=q0.w;
  out[4]=q1.x; out[5]=q1.y; out[6]=q1.z; out[7]=q1.w;
}

__device__ __forceinline__ void load8_sum(float out[8], int k0,
    const float* __restrict__ A0, int r0,
    const float* __restrict__ A1, int r1, int K, float s0, float s1)
{
  const float4* p0 = (const float4*)(A0 + (size_t)r0 * K + k0);
  float4 a0 = p0[0], a1 = p0[1];
  float4 b0 = make_float4(0.f,0.f,0.f,0.f), b1 = b0;
  if (r1 >= 0) {
    const float4* p1 = (const float4*)(A1 + (size_t)r1 * K + k0);
    b0 = p1[0]; b1 = p1[1];
  }
  out[0]=a0.x*s0+b0.x*s1; out[1]=a0.y*s0+b0.y*s1;
  out[2]=a0.z*s0+b0.z*s1; out[3]=a0.w*s0+b0.w*s1;
  out[4]=a1.x*s0+b1.x*s1; out[5]=a1.y*s0+b1.y*s1;
  out[6]=a1.z*s0+b1.z*s1; out[7]=a1.w*s0+b1.w*s1;
}

// ---------------------------------------------------------------------------
// Gather-GEMM: D = act(Cadd + A @ B + bias).
// MODE 0: A = concat(A0[g0[row]] (K0), A1[g1[row]] (K1))  (gathers optional)
// MODE 1: A = A0[g0[row]]*s0 + A1[g1[row]]*s1, width K0; g1 may yield -1 (zero row)
// One wave computes a 16x128 tile: 8x v_wmma_f32_16x16x32_bf16 per K-step,
// A fragment reused across all 8 N-subtiles; B read from packed bf16 fragments.
// Block = 256 threads = 8 waves.
// ---------------------------------------------------------------------------
template<int MODE>
__global__ __launch_bounds__(256)
void wmma_gemm_kernel(const float* __restrict__ A0, const int* __restrict__ g0, int K0,
                      const float* __restrict__ A1, const int* __restrict__ g1, int K1,
                      float s0, float s1,
                      const bf16_t* __restrict__ Bp, const float* __restrict__ bias,
                      const float* __restrict__ Cadd,
                      float* __restrict__ D, float* __restrict__ D2,
                      int M, int N, int act)
{
  const int lane   = threadIdx.x & 31;
  const int wid    = threadIdx.x >> 5;
  const int nt128  = N >> 7;
  const int mtiles = (M + 15) >> 4;
  const int tile   = blockIdx.x * 8 + wid;
  if (tile >= mtiles * nt128) return;
  const int mt = tile / nt128;
  const int nt = tile % nt128;

  const int Ktot = (MODE == 1) ? K0 : (K0 + K1);
  const int Kt32 = Ktot >> 5;

  const int mrow  = (mt << 4) + (lane & 15);
  const int mrowc = mrow < M ? mrow : (M - 1);
  int r0, r1 = -1;
  r0 = g0 ? g0[mrowc] : mrowc;
  if (MODE == 1)      r1 = g1 ? g1[mrowc] : mrowc;
  else if (A1)        r1 = g1 ? g1[mrowc] : mrowc;

  // A 16x32 bf16 layout: lanes 0-15 hold M=lane, K 0-7 & 16-23; lanes 16-31 K 8-15 & 24-31.
  const int ka = (lane >> 4) * 8;

  v8f acc[8] = {};

  const bf16_t* bbase = Bp + ((size_t)(nt * 8) * Kt32) * 512 + (size_t)lane * 16;

  for (int kt32 = 0; kt32 < Kt32; ++kt32) {
    const int kt = kt32 << 5;
    float alo[8], ahi[8];
    if (MODE == 0) {
      load8_concat(alo, kt + ka,      A0, r0, K0, A1, r1, K1);
      load8_concat(ahi, kt + 16 + ka, A0, r0, K0, A1, r1, K1);
    } else {
      load8_sum(alo, kt + ka,      A0, r0, A1, r1, K0, s0, s1);
      load8_sum(ahi, kt + 16 + ka, A0, r0, A1, r1, K0, s0, s1);
    }
    v16bf af;
    #pragma unroll
    for (int j = 0; j < 8; ++j) { af[j] = (bf16_t)alo[j]; af[8 + j] = (bf16_t)ahi[j]; }

    #pragma unroll
    for (int sub = 0; sub < 8; ++sub) {
      const v16bf bfv = *(const v16bf*)(bbase + ((size_t)sub * Kt32 + kt32) * 512);
      acc[sub] = __builtin_amdgcn_wmma_f32_16x16x32_bf16(
          false, af, false, bfv, (short)0, acc[sub], false, false);
    }
  }

  // C/D layout: lanes 0-15 hold rows 0-7 (vgpr r -> row r), lanes 16-31 rows 8-15.
  const int mbase = (mt << 4) + (lane >> 4) * 8;
  const int nbase = (nt << 7);
  const int nloc  = lane & 15;
  #pragma unroll
  for (int sub = 0; sub < 8; ++sub) {
    const int gcol = nbase + sub * 16 + nloc;
    const float bv = bias ? bias[gcol] : 0.f;
    #pragma unroll
    for (int r = 0; r < 8; ++r) {
      const int grow = mbase + r;
      if (grow < M) {
        float v = acc[sub][r] + bv;
        if (Cadd) v += Cadd[(size_t)grow * N + gcol];
        v = act_apply(v, act);
        D[(size_t)grow * N + gcol] = v;
        if (D2) D2[(size_t)grow * N + gcol] = v;
      }
    }
  }
}

// ---- reverse-edge index via open-addressing hash (keys unique: u*N_NODES+v) ----
__global__ void hash_insert_kernel(const int* __restrict__ src, const int* __restrict__ dst,
                                   int E, int* __restrict__ hkey, int* __restrict__ hval)
{
  int e = blockIdx.x * blockDim.x + threadIdx.x;
  if (e >= E) return;
  int key = src[e] * N_NODES + dst[e];
  unsigned h = ((unsigned)key * 2654435761u) >> (32 - HASH_BITS);
  for (;;) {
    int* slot = &hkey[h & HASH_MASK];
    int prev = atomicCAS(slot, -1, key);
    if (prev == -1 || prev == key) { hval[h & HASH_MASK] = e; break; }
    ++h;
  }
}

__global__ void hash_lookup_kernel(const int* __restrict__ src, const int* __restrict__ dst,
                                   int E, const int* __restrict__ hkey,
                                   const int* __restrict__ hval, int* __restrict__ rev)
{
  int e = blockIdx.x * blockDim.x + threadIdx.x;
  if (e >= E) return;
  int key = dst[e] * N_NODES + src[e];
  unsigned h = ((unsigned)key * 2654435761u) >> (32 - HASH_BITS);
  int r = -1;
  for (;;) {
    int k = hkey[h & HASH_MASK];
    if (k == key) { r = hval[h & HASH_MASK]; break; }
    if (k == -1) break;
    ++h;
  }
  rev[e] = r;
}

// segment_sum over dst, float4 per thread: out[dst[e]][c..c+3] += eh[e][c..c+3]
__global__ void scatter_add4_kernel(const float* __restrict__ eh, const int* __restrict__ dst,
                                    float* __restrict__ out, int E, int hshift)
{
  int i = blockIdx.x * blockDim.x + threadIdx.x;       // over E * H/4
  int total = E << (hshift - 2);
  if (i >= total) return;
  int e  = i >> (hshift - 2);
  int c4 = (i & ((1 << (hshift - 2)) - 1)) << 2;
  const float4 v = *(const float4*)(eh + ((size_t)e << hshift) + c4);
  float* o = out + ((size_t)dst[e] << hshift) + c4;
  atomicAdd(o + 0, v.x); atomicAdd(o + 1, v.y);
  atomicAdd(o + 2, v.z); atomicAdd(o + 3, v.w);
}

// ---- graph layernorm (mode='graph', biased var, eps=1e-5) + elu ----
__global__ void ln_stats_kernel(const float* __restrict__ h, const int* __restrict__ batch,
                                int n_nodes, int F, float* gs, float* gq, float* gc)
{
  int n = blockIdx.x * blockDim.x + threadIdx.x;
  if (n >= n_nodes) return;
  float s = 0.f, q = 0.f;
  const float* row = h + (size_t)n * F;
  for (int c = 0; c < F; ++c) { float v = row[c]; s += v; q += v * v; }
  int g = batch[n];
  atomicAdd(&gs[g], s); atomicAdd(&gq[g], q); atomicAdd(&gc[g], 1.f);
}

__global__ void ln_final_kernel(float* gs, float* gq, const float* gc, int F)
{
  int g = threadIdx.x;
  if (g >= NUM_G) return;
  float norm = fmaxf(gc[g], 1.f) * (float)F;
  float mean = gs[g] / norm;
  float var  = gq[g] / norm - mean * mean;
  gs[g] = mean;
  gq[g] = rsqrtf(var + LN_EPS);
}

__global__ void ln_apply_kernel(const float* __restrict__ h, const int* __restrict__ batch,
                                const float* __restrict__ w, const float* __restrict__ b,
                                const float* __restrict__ mean, const float* __restrict__ rstd,
                                float* __restrict__ out, int n_nodes, int F)
{
  int i = blockIdx.x * blockDim.x + threadIdx.x;
  if (i >= n_nodes * F) return;
  int n = i / F; int c = i - n * F;
  int g = batch[n];
  float v = (h[i] - mean[g]) * rstd[g] * w[c] + b[c];
  out[i] = v > 0.f ? v : (__expf(v) - 1.f);
}

extern "C" void kernel_launch(void* const* d_in, const int* in_sizes, int n_in,
                              void* d_out, int out_size, void* d_ws, size_t ws_size,
                              hipStream_t stream)
{
  const float* x         = (const float*)d_in[0];
  const float* edge_attr = (const float*)d_in[1];
  const int*   ei        = (const int*)d_in[2];
  const int*   batch     = (const int*)d_in[3];
  const float* lin_up_w  = (const float*)d_in[4];
  const float* lin_up_b  = (const float*)d_in[5];
  const float* lin_up2_w = (const float*)d_in[6];
  const float* lin_up2_b = (const float*)d_in[7];
  const float* c1_edge_w = (const float*)d_in[8];
  const float* c1_edge_b = (const float*)d_in[9];
  const float* c1_msg_w  = (const float*)d_in[10];
  const float* c1_msg_b  = (const float*)d_in[11];
  const float* c1_node_w = (const float*)d_in[12];
  const float* c1_node_b = (const float*)d_in[13];
  const float* c2_edge_w = (const float*)d_in[14];
  const float* c2_edge_b = (const float*)d_in[15];
  const float* c2_msg_w  = (const float*)d_in[16];
  const float* c2_msg_b  = (const float*)d_in[17];
  const float* c2_node_w = (const float*)d_in[18];
  const float* c2_node_b = (const float*)d_in[19];
  const float* norm1_w   = (const float*)d_in[20];
  const float* norm1_b   = (const float*)d_in[21];
  const float* norm2_w   = (const float*)d_in[22];
  const float* norm2_b   = (const float*)d_in[23];

  const int E = in_sizes[2] / 2;
  const int* src = ei;
  const int* dst = ei + E;

  // ---- carve workspace ----
  size_t off = 0;
  auto carve = [&](size_t bytes) -> void* {
    void* p = (char*)d_ws + off;
    off += (bytes + 255) & ~(size_t)255;
    return p;
  };
  int*   rev   = (int*)  carve((size_t)E * 4);
  int*   hkey  = (int*)  carve((size_t)HASH_SZ * 4);
  int*   hval  = (int*)  carve((size_t)HASH_SZ * 4);
  float* ea    = (float*)carve((size_t)E * F_E * 4);
  float* eh0   = (float*)carve((size_t)E * F_IN * 4);   // reused at [E,128] for conv2
  float* ehA   = (float*)carve((size_t)E * F_IN * 4);
  float* ehB   = (float*)carve((size_t)E * F_IN * 4);
  float* insum = (float*)carve((size_t)N_NODES * F_IN * 4);
  float* hbuf  = (float*)carve((size_t)N_NODES * F_IN * 4);
  float* x1    = (float*)carve((size_t)N_NODES * F_IN * 4);
  float* gst   = (float*)carve(256 * 4);
  float* gs = gst; float* gq = gst + 64; float* gc = gst + 128;
  // packed bf16 weight fragments
  bf16_t* pk_up1  = (bf16_t*)carve((size_t)F_E * F_E * 2);
  bf16_t* pk_up2  = (bf16_t*)carve((size_t)F_E * F_E * 2);
  bf16_t* pk_c1e  = (bf16_t*)carve((size_t)(F_IN + F_E) * F_IN * 2);
  bf16_t* pk_c1m  = (bf16_t*)carve((size_t)F_IN * F_IN * 2);
  bf16_t* pk_c1n  = (bf16_t*)carve((size_t)(F_IN + F_IN) * F_IN * 2);
  bf16_t* pk_c2e  = (bf16_t*)carve((size_t)(F_IN + F_E) * F_OUT * 2);
  bf16_t* pk_c2m  = (bf16_t*)carve((size_t)F_OUT * F_OUT * 2);
  bf16_t* pk_c2n  = (bf16_t*)carve((size_t)(F_IN + F_OUT) * F_OUT * 2);

  float* x2_out  = (float*)d_out;                               // [N_NODES, 128]
  float* ea2_out = (float*)d_out + (size_t)N_NODES * F_OUT;     // [E, 128]

  auto pack = [&](const float* B, bf16_t* P, int K, int N) {
    int total = (N >> 4) * (K >> 5) * 32;
    pack_b_kernel<<<(total + 255) / 256, 256, 0, stream>>>(B, P, K, N);
  };

  auto gemm = [&](bool sum_mode, const float* A0, const int* g0, int K0,
                  const float* A1, const int* g1, int K1, float sc0, float sc1,
                  const bf16_t* Bp, const float* bias, const float* Cadd,
                  float* D, float* D2, int M, int N, int act) {
    int tiles  = ((M + 15) >> 4) * (N >> 7);
    int blocks = (tiles + 7) >> 3;
    if (sum_mode)
      wmma_gemm_kernel<1><<<blocks, 256, 0, stream>>>(A0, g0, K0, A1, g1, K1, sc0, sc1,
                                                      Bp, bias, Cadd, D, D2, M, N, act);
    else
      wmma_gemm_kernel<0><<<blocks, 256, 0, stream>>>(A0, g0, K0, A1, g1, K1, sc0, sc1,
                                                      Bp, bias, Cadd, D, D2, M, N, act);
  };

  const int eb = (E + 255) / 256;

  // ---- pack all weights into bf16 WMMA fragments ----
  pack(lin_up_w,  pk_up1, F_E, F_E);
  pack(lin_up2_w, pk_up2, F_E, F_E);
  pack(c1_edge_w, pk_c1e, F_IN + F_E, F_IN);
  pack(c1_msg_w,  pk_c1m, F_IN, F_IN);
  pack(c1_node_w, pk_c1n, F_IN + F_IN, F_IN);
  pack(c2_edge_w, pk_c2e, F_IN + F_E, F_OUT);
  pack(c2_msg_w,  pk_c2m, F_OUT, F_OUT);
  pack(c2_node_w, pk_c2n, F_IN + F_OUT, F_OUT);

  // ---- reverse-edge index ----
  hipMemsetAsync(hkey, 0xFF, (size_t)HASH_SZ * 4, stream);
  hash_insert_kernel<<<eb, 256, 0, stream>>>(src, dst, E, hkey, hval);
  hash_lookup_kernel<<<eb, 256, 0, stream>>>(src, dst, E, hkey, hval, rev);

  // ---- ea = elu(edge_attr @ lin_up_w + b) ----
  gemm(false, edge_attr, nullptr, F_E, nullptr, nullptr, 0, 1.f, 1.f,
       pk_up1, lin_up_b, nullptr, ea, nullptr, E, F_E, ACT_ELU);

  // ---- conv1 (D-MPNN, H=512) ----
  gemm(false, x, src, F_IN, ea, nullptr, F_E, 1.f, 1.f,
       pk_c1e, c1_edge_b, nullptr, eh0, ehA, E, F_IN, ACT_RELU);
  {
    float* cur = ehA; float* nxt = ehB;
    const int sb = ((E << 7) + 255) / 256;   // E * 512/4 threads
    for (int t = 0; t < T_STEPS; ++t) {
      hipMemsetAsync(insum, 0, (size_t)N_NODES * F_IN * 4, stream);
      scatter_add4_kernel<<<sb, 256, 0, stream>>>(cur, dst, insum, E, 9);
      // eh' = relu(eh0 + (in_sum[src] - eh[rev]) @ Wm + bm), fused gather-sum A
      gemm(true, insum, src, F_IN, cur, rev, F_IN, 1.f, -1.f,
           pk_c1m, c1_msg_b, eh0, nxt, nullptr, E, F_IN, ACT_RELU);
      float* tmp = cur; cur = nxt; nxt = tmp;
    }
    hipMemsetAsync(insum, 0, (size_t)N_NODES * F_IN * 4, stream);
    scatter_add4_kernel<<<sb, 256, 0, stream>>>(cur, dst, insum, E, 9);
  }
  gemm(false, x, nullptr, F_IN, insum, nullptr, F_IN, 1.f, 1.f,
       pk_c1n, c1_node_b, nullptr, hbuf, nullptr, N_NODES, F_IN, ACT_RELU);

  // ---- layernorm1 + elu -> x1 ----
  hipMemsetAsync(gst, 0, 256 * 4, stream);
  ln_stats_kernel<<<(N_NODES + 255) / 256, 256, 0, stream>>>(hbuf, batch, N_NODES, F_IN, gs, gq, gc);
  ln_final_kernel<<<1, 64, 0, stream>>>(gs, gq, gc, F_IN);
  ln_apply_kernel<<<((N_NODES * F_IN) + 255) / 256, 256, 0, stream>>>(
      hbuf, batch, norm1_w, norm1_b, gs, gq, x1, N_NODES, F_IN);

  // ---- ea2 = elu(ea @ lin_up2_w + b) -> output slice 2 ----
  gemm(false, ea, nullptr, F_E, nullptr, nullptr, 0, 1.f, 1.f,
       pk_up2, lin_up2_b, nullptr, ea2_out, nullptr, E, F_E, ACT_ELU);

  // ---- conv2 (D-MPNN, H=128), reuse eh0/ehA/ehB/insum ----
  gemm(false, x1, src, F_IN, ea2_out, nullptr, F_E, 1.f, 1.f,
       pk_c2e, c2_edge_b, nullptr, eh0, ehA, E, F_OUT, ACT_RELU);
  {
    float* cur = ehA; float* nxt = ehB;
    const int sb = ((E << 5) + 255) / 256;   // E * 128/4 threads
    for (int t = 0; t < T_STEPS; ++t) {
      hipMemsetAsync(insum, 0, (size_t)N_NODES * F_OUT * 4, stream);
      scatter_add4_kernel<<<sb, 256, 0, stream>>>(cur, dst, insum, E, 7);
      gemm(true, insum, src, F_OUT, cur, rev, F_OUT, 1.f, -1.f,
           pk_c2m, c2_msg_b, eh0, nxt, nullptr, E, F_OUT, ACT_RELU);
      float* tmp = cur; cur = nxt; nxt = tmp;
    }
    hipMemsetAsync(insum, 0, (size_t)N_NODES * F_OUT * 4, stream);
    scatter_add4_kernel<<<sb, 256, 0, stream>>>(cur, dst, insum, E, 7);
  }
  gemm(false, x1, nullptr, F_IN, insum, nullptr, F_OUT, 1.f, 1.f,
       pk_c2n, c2_node_b, nullptr, hbuf, nullptr, N_NODES, F_OUT, ACT_RELU);

  // ---- layernorm2 + elu -> x2 (output slice 1) ----
  hipMemsetAsync(gst, 0, 256 * 4, stream);
  ln_stats_kernel<<<(N_NODES + 255) / 256, 256, 0, stream>>>(hbuf, batch, N_NODES, F_OUT, gs, gq, gc);
  ln_final_kernel<<<1, 64, 0, stream>>>(gs, gq, gc, F_OUT);
  ln_apply_kernel<<<((N_NODES * F_OUT) + 255) / 256, 256, 0, stream>>>(
      hbuf, batch, norm2_w, norm2_b, gs, gq, x2_out, N_NODES, F_OUT);
}
